// LengthRegulator_51084341019265
// MI455X (gfx1250) — matrix operand order repaired and checked by reading, not compiled, not run on hardware
//
#include <hip/hip_runtime.h>

// ---------------------------------------------------------------------------
// LengthRegulator  out[b,d,m] = sum_n x[b,d,n] * path[b,n,m]
// path is one-hot over n per (b,m)  =>  out[b,d,m] = x[b,d, idx[b,m]]
//
// Phase 1: idx[b,m] = sum_n n*path[b,n,m] computed on the matrix unit
//          (v_wmma_f32_16x16x32_f16, exact: n<=511 is exact in f16,
//           path in {0,1}, f32 accumulation). Streams path (268 MB) once
//          with non-temporal scalar f32 loads. Iota A-matrix is hoisted and
//          advanced by packed f16 adds (+32 per k-step, always exact).
// Phase 2: LDS-staged gather + non-temporal b128 stores of out (201 MB).
// Total HBM traffic ~494 MB -> ~21 us at 23.3 TB/s; exact fp32 output.
// ---------------------------------------------------------------------------

typedef __attribute__((ext_vector_type(16))) _Float16 v16h;
typedef __attribute__((ext_vector_type(8)))  float    v8f;
typedef __attribute__((ext_vector_type(4)))  float    v4f;
typedef __attribute__((ext_vector_type(4)))  int      v4i;

#define BB 32
#define DD 384
#define TT 512     // T_text (K), 16 WMMA k-steps of 32
#define TM 4096    // T_mel

// ---------------------------------------------------------------------------
// Phase 1: one wave per (b, 16-mel-column tile). 32*256 = 8192 waves.
// D[i][j] = sum_k A[i][k]*B[k][j] with A[i][k] = k (iota, independent of i)
//   => every row of D equals idx[m0..m0+15]; read it from acc[0], lanes 0..15.
// A layout (ISA 7.12.2, 16-bit A 16x32): lane half h=lane>>4,
//   element e<8 -> K = 8h+e ; e>=8 -> K = 8h+e+8.
// B layout: N = lane&15, element e -> same K as A (path row k0+K).
//   => lane loads 16 strided floats, rows k0+K of mel column m0+N: 16 scalar loads.
// ---------------------------------------------------------------------------
__global__ __launch_bounds__(256) void lr_idx_wmma(const float* __restrict__ path,
                                                   int* __restrict__ idx) {
  const int wid  = blockIdx.x * (blockDim.x >> 5) + (threadIdx.x >> 5);
  const int lane = threadIdx.x & 31;
  const int b    = wid >> 8;      // 256 m-tiles per batch
  const int m0   = (wid & 255) << 4;

  // Iota A-matrix for k0 = 0; advanced by +32 per k-step (exact in f16).
  const int h = (lane >> 4) & 1;
  v16h amat;
#pragma unroll
  for (int e = 0; e < 16; ++e) {
    const int k = (e < 8) ? (8 * h + e) : (8 * h + e + 8);
    amat[e] = (_Float16)(float)k;
  }
  const _Float16 kstep = (_Float16)32.0f;

  const float* pbase = path + (size_t)b * TT * TM + m0;
  v8f acc = {};

#pragma unroll 4
  for (int k0 = 0; k0 < TT; k0 += 32) {
    // B: this lane holds mel column m0+(lane&15), element e -> path row k0+K (same K as A)
    const float* prow = pbase + (lane & 15);
    v16h bmat;
#pragma unroll
    for (int e = 0; e < 16; ++e) {
      const int k = (e < 8) ? (8 * h + e) : (8 * h + e + 8);
      bmat[e] = (_Float16)__builtin_nontemporal_load(prow + (size_t)(k0 + k) * TM);
    }
    // (neg_a, A, neg_b, B, c_mod, C, reuse_a, reuse_b)
    acc = __builtin_amdgcn_wmma_f32_16x16x32_f16(false, amat, false, bmat,
                                                 (short)0, acc, false, false);
    // Advance iota: packed f16 adds, all values stay integer-exact (< 512)
#pragma unroll
    for (int e = 0; e < 16; ++e) amat[e] += kstep;
  }

  // C/D layout: VGPR0, lanes 0..15 -> (M=0, N=lane); D is constant over M.
  if (lane < 16) idx[b * TM + m0 + lane] = (int)acc[0];
}

// ---------------------------------------------------------------------------
// Phase 2: block per (b,d) row. Stage x row (2 KB) in LDS, gather via sorted
// idx (int4 coalesced, L2-resident), stream out with non-temporal b128.
// ---------------------------------------------------------------------------
__global__ __launch_bounds__(256) void lr_expand(const float* __restrict__ x,
                                                 const int* __restrict__ idx,
                                                 float* __restrict__ out) {
  __shared__ float xs[TT];
  const int bd  = blockIdx.x;          // b*DD + d
  const int b   = bd / DD;
  const int tid = threadIdx.x;

  const float* xrow = x + (size_t)bd * TT;
#pragma unroll
  for (int i = tid; i < TT; i += 256) xs[i] = xrow[i];
  __syncthreads();

  const int* irow = idx + b * TM;
  float*     orow = out + (size_t)bd * TM;
#pragma unroll
  for (int m = tid * 4; m < TM; m += 256 * 4) {
    v4i i4 = *(const v4i*)(irow + m);
    v4f o  = { xs[i4[0]], xs[i4[1]], xs[i4[2]], xs[i4[3]] };
    __builtin_nontemporal_store(o, (v4f*)(orow + m));
  }
}

// ---------------------------------------------------------------------------
extern "C" void kernel_launch(void* const* d_in, const int* in_sizes, int n_in,
                              void* d_out, int out_size, void* d_ws, size_t ws_size,
                              hipStream_t stream) {
  const float* x    = (const float*)d_in[0];   // [B, D, T_text] fp32
  const float* path = (const float*)d_in[1];   // [B, T_text, T_mel] fp32 one-hot
  float*       out  = (float*)d_out;           // [B, D, T_mel] fp32
  int*         idx  = (int*)d_ws;              // [B, T_mel] scratch (512 KB)

  // Phase 1: 8192 waves = 1024 blocks x 8 waves (256 threads)
  lr_idx_wmma<<<dim3((BB * (TM / 16)) / 8), dim3(256), 0, stream>>>(path, idx);
  // Phase 2: one block per (b,d) row
  lr_expand<<<dim3(BB * DD), dim3(256), 0, stream>>>(x, idx, out);
}
